// Gemma3Attention_13288628814511
// MI455X (gfx1250) — matrix operand order repaired and checked by reading, not compiled, not run on hardware
//
#include <hip/hip_runtime.h>
#include <hip/hip_bf16.h>

#define B_      2
#define S_      4096
#define HID_    640
#define NH_     4
#define HD_     256
#define QKVN_   1536          // NH*HD + HD + HD
#define NQD_    1024          // NH*HD
#define WINDOW_ 512
#define EPS_    1e-6f
#define SCALING_ 0.0625f      // 256^-0.5
#define SOFTCAP_ 50.0f

typedef __attribute__((ext_vector_type(16))) _Float16 v16h;
typedef __attribute__((ext_vector_type(8)))  _Float16 v8h;
typedef __attribute__((ext_vector_type(8)))  float    v8f;
typedef __attribute__((ext_vector_type(4)))  unsigned int u32x4;
typedef __attribute__((ext_vector_type(4)))  int i32x4;
typedef __attribute__((ext_vector_type(8)))  int i32x8;

#if __has_builtin(__builtin_amdgcn_tensor_load_to_lds)
#define HAVE_TDM 1
#else
#define HAVE_TDM 0
#endif

// ---------------------------------------------------------------------------
// TDM: 2D tile load global->LDS.  D# per CDNA5 ISA.
//   g0: [0]=count=1, [1]=lds_addr(bytes), [2]=gaddr[31:0],
//       [3]=gaddr[56:32] | type=2<<30
//   g1: [0]=data_size(2B)<<16, tensor_dim0@79:48, tensor_dim1@111:80,
//       tile_dim0@127:112, tile_dim1@143:128, dim0_stride@207:160
// This toolchain exposes the 6-arg builtin:
//   (u32x4 g0, i32x8 g1, i32x4 g2, i32x4 g3, i32x8 extra, i32 cpol)
// ---------------------------------------------------------------------------
#if HAVE_TDM
__device__ __forceinline__ void tdm_load_2d(unsigned lds_off, const void* gptr,
                                            unsigned d0_elems, unsigned d1_rows,
                                            unsigned stride_elems) {
  unsigned long long ga = (unsigned long long)gptr;
  u32x4 g0;
  g0[0] = 1u;                                   // count=1, user descriptor
  g0[1] = lds_off;                              // LDS byte address
  g0[2] = (unsigned)(ga & 0xFFFFFFFFu);
  g0[3] = (unsigned)((ga >> 32) & 0x01FFFFFFu) | (2u << 30);   // type=2
  i32x8 g1;
  g1[0] = (int)(1u << 16);                      // data_size = 2 bytes
  g1[1] = (int)((d0_elems & 0xFFFFu) << 16);    // tensor_dim0[15:0]
  g1[2] = (int)(((d0_elems >> 16) & 0xFFFFu) | ((d1_rows & 0xFFFFu) << 16));
  g1[3] = (int)(((d1_rows >> 16) & 0xFFFFu) | ((d0_elems & 0xFFFFu) << 16)); // tile_dim0
  g1[4] = (int)(d1_rows & 0xFFFFu);             // tile_dim1 (tile_dim2 = 0)
  g1[5] = (int)stride_elems;                    // dim0_stride[31:0]
  g1[6] = 0;
  g1[7] = 0;
  i32x4 gz4 = {0, 0, 0, 0};
  i32x8 gz8 = {0, 0, 0, 0, 0, 0, 0, 0};
  __builtin_amdgcn_tensor_load_to_lds(g0, g1, gz4, gz4, gz8, 0);
}
__device__ __forceinline__ void tdm_wait() {
#if __has_builtin(__builtin_amdgcn_s_wait_tensorcnt)
  __builtin_amdgcn_s_wait_tensorcnt(0);
#else
  asm volatile("s_wait_tensorcnt 0" ::: "memory");
#endif
}
#endif

// ---------------------------------------------------------------------------
// WMMA helpers (gfx1250 wave32), layouts per CDNA5 ISA.
// ---------------------------------------------------------------------------
__device__ __forceinline__ v8f wmma_f16(v16h a, v16h b, v8f c) {
  return __builtin_amdgcn_wmma_f32_16x16x32_f16(false, a, false, b,
                                                (short)0, c, false, false);
}

// A 16x32 tile, row-major, row stride lda (halves). base = &A[row0][k0].
__device__ __forceinline__ v16h load_a_frag(const _Float16* base, int lda, int lane) {
  const int m  = lane & 15;
  const int kb = (lane >> 4) * 8;
  const _Float16* p = base + (size_t)m * lda + kb;
  v8h lo = *(const v8h*)(p);
  v8h hi = *(const v8h*)(p + 16);
  return __builtin_shufflevector(lo, hi, 0,1,2,3,4,5,6,7,8,9,10,11,12,13,14,15);
}

// B 32x16 tile, column-major storage, column stride ldb. colbase = &Bt[n0][k0].
__device__ __forceinline__ v16h load_b_frag(const _Float16* colbase, int ldb, int lane) {
  const int n  = lane & 15;
  const int kb = (lane >> 4) * 8;
  const _Float16* p = colbase + (size_t)n * ldb + kb;
  v8h lo = *(const v8h*)(p);
  v8h hi = *(const v8h*)(p + 16);
  return __builtin_shufflevector(lo, hi, 0,1,2,3,4,5,6,7,8,9,10,11,12,13,14,15);
}

// ---------------------------------------------------------------------------
// Conversion / packing
// ---------------------------------------------------------------------------
__global__ void __launch_bounds__(256)
cvt_f32_to_f16(const float* __restrict__ s, _Float16* __restrict__ d, int n) {
  int i = blockIdx.x * blockDim.x + threadIdx.x;
  int stride = gridDim.x * blockDim.x;
  for (; i < n; i += stride) d[i] = (_Float16)s[i];
}

__global__ void __launch_bounds__(256)
pack_wqkv(const float* __restrict__ Wq, const float* __restrict__ Wk,
          const float* __restrict__ Wv, _Float16* __restrict__ Wt) {
  const int total = QKVN_ * HID_;
  int i = blockIdx.x * blockDim.x + threadIdx.x;
  int stride = gridDim.x * blockDim.x;
  for (; i < total; i += stride) {
    int n = i / HID_, k = i % HID_;
    float v;
    if (n < NQD_)            v = Wq[(size_t)k * NQD_ + n];
    else if (n < NQD_ + HD_) v = Wk[(size_t)k * HD_ + (n - NQD_)];
    else                     v = Wv[(size_t)k * HD_ + (n - NQD_ - HD_)];
    Wt[i] = (_Float16)v;
  }
}

__global__ void __launch_bounds__(256)
pack_wo(const float* __restrict__ Wo, _Float16* __restrict__ Wt) {
  const int total = HID_ * NQD_;
  int i = blockIdx.x * blockDim.x + threadIdx.x;
  int stride = gridDim.x * blockDim.x;
  for (; i < total; i += stride) {
    int n = i / NQD_, k = i % NQD_;
    Wt[i] = (_Float16)Wo[(size_t)k * HID_ + n];
  }
}

// ---------------------------------------------------------------------------
// WMMA GEMM: C[M,N] = A[M,K] x Bt[N,K].  A panel staged via TDM into LDS.
// grid.x = M/16, grid.y = N/128; block = 128 (4 waves x 2 N-tiles)
// ---------------------------------------------------------------------------
__global__ void __launch_bounds__(128)
gemm16_kernel(const _Float16* __restrict__ A, const _Float16* __restrict__ Bt,
              int K, float* __restrict__ C, int N) {
  __shared__ _Float16 ldsA[16 * 1024];          // up to K=1024
  const int lane = threadIdx.x & 31;
  const int wave = threadIdx.x >> 5;
  const int m0 = blockIdx.x * 16;
  const int n0 = blockIdx.y * 128 + wave * 32;
  const _Float16* Abase = A + (size_t)m0 * K;

#if HAVE_TDM
  if (wave == 0) {
    tdm_load_2d((unsigned)(size_t)(void*)ldsA, Abase,
                (unsigned)K, 16u, (unsigned)K);
    tdm_wait();
  }
#else
  for (int i = threadIdx.x * 8; i < 16 * K; i += 128 * 8)
    *(v8h*)(ldsA + i) = *(const v8h*)(Abase + i);
#endif
  __syncthreads();

  v8f acc0 = {0,0,0,0,0,0,0,0};
  v8f acc1 = {0,0,0,0,0,0,0,0};
  for (int k0 = 0; k0 < K; k0 += 32) {
    v16h af = load_a_frag(ldsA + k0, K, lane);
    v16h b0 = load_b_frag(Bt + (size_t)n0 * K + k0, K, lane);
    v16h b1 = load_b_frag(Bt + (size_t)(n0 + 16) * K + k0, K, lane);
    acc0 = wmma_f16(af, b0, acc0);
    acc1 = wmma_f16(af, b1, acc1);
  }
  const int n = lane & 15, hi = lane >> 4;
#pragma unroll
  for (int r = 0; r < 8; ++r) {
    const size_t m = (size_t)(m0 + r + hi * 8);
    C[m * N + n0 + n]      = acc0[r];
    C[m * N + n0 + 16 + n] = acc1[r];
  }
}

// ---------------------------------------------------------------------------
// RMS-norm + RoPE epilogue
// ---------------------------------------------------------------------------
__global__ void __launch_bounds__(192)
qkv_post_kernel(const float* __restrict__ qkv, const float* __restrict__ cosb,
                const float* __restrict__ sinb, const float* __restrict__ qw,
                const float* __restrict__ kw, _Float16* __restrict__ qf,
                _Float16* __restrict__ kf, _Float16* __restrict__ vt) {
  const int bs = blockIdx.x;
  const int b = bs / S_, s = bs % S_;
  const int wave = threadIdx.x >> 5;
  const int lane = threadIdx.x & 31;
  const float* row = qkv + (size_t)bs * QKVN_;

  if (wave == 5) {
    const float* v = row + NQD_ + HD_;
#pragma unroll
    for (int i = 0; i < 8; ++i) {
      int d = lane * 8 + i;
      vt[((size_t)(b * HD_ + d)) * S_ + s] = (_Float16)v[d];
    }
    return;
  }
  const float* x;
  const float* w;
  if (wave < 4) { x = row + wave * HD_; w = qw; }
  else          { x = row + NQD_;       w = kw; }

  const int d0 = lane * 8;
  float xv[8];
  float ss = 0.f;
#pragma unroll
  for (int i = 0; i < 8; ++i) { xv[i] = x[d0 + i]; ss += xv[i] * xv[i]; }
#pragma unroll
  for (int xm = 1; xm < 32; xm <<= 1) ss += __shfl_xor(ss, xm, 32);
  const float rinv = rsqrtf(ss * (1.0f / HD_) + EPS_);

  const int dp = (d0 < HD_ / 2) ? d0 + HD_ / 2 : d0 - HD_ / 2;
  const float sgn = (d0 < HD_ / 2) ? -1.f : 1.f;
  const float* cr = cosb + (size_t)s * HD_;
  const float* sr = sinb + (size_t)s * HD_;
  _Float16 outv[8];
#pragma unroll
  for (int i = 0; i < 8; ++i) {
    int d = d0 + i;
    float xn = xv[i]     * rinv * (1.f + w[d]);
    float pn = x[dp + i] * rinv * (1.f + w[dp + i]);
    outv[i] = (_Float16)(xn * cr[d] + sgn * pn * sr[d]);
  }
  if (wave < 4) {
    _Float16* dst = qf + (((size_t)b * NH_ + wave) * S_ + s) * HD_ + d0;
#pragma unroll
    for (int i = 0; i < 8; ++i) dst[i] = outv[i];
  } else {
    _Float16* dst = kf + ((size_t)b * S_ + s) * HD_ + d0;
#pragma unroll
    for (int i = 0; i < 8; ++i) dst[i] = outv[i];
  }
}

// ---------------------------------------------------------------------------
// Sliding-window flash attention, block-cooperative K/V staging via TDM.
// Block = 4 waves = 4 consecutive 16-query tiles of one (b,h); uniform chunk
// loop over the union window; K/V chunks shared through LDS.
// grid = B*NH*(S/64) = 512 blocks x 128 threads.
// ---------------------------------------------------------------------------
__global__ void __launch_bounds__(128)
attn_kernel(const _Float16* __restrict__ qf, const _Float16* __restrict__ kf,
            const _Float16* __restrict__ vt, _Float16* __restrict__ ao) {
  // smem: K chunk 32x256 (16KB) | V chunk 256x32 (16KB) | P tiles 4x16x32 (4KB)
  __shared__ _Float16 smem[8192 + 8192 + 4 * 512];
  _Float16* Klds = smem;
  _Float16* Vlds = smem + 8192;
  const int lane = threadIdx.x & 31;
  const int wave = threadIdx.x >> 5;
  _Float16* P = smem + 16384 + wave * 512;

  const int blk = blockIdx.x;
  const int qt64 = blk & (S_ / 64 - 1);
  const int h = (blk >> 6) & (NH_ - 1);
  const int b = blk >> 8;
  const int qb0 = qt64 * 64;
  const int qr0 = qb0 + wave * 16;
  const int hi = lane >> 4, ln = lane & 15;

  const _Float16* Q  = qf + (((size_t)b * NH_ + h) * S_ + qr0) * HD_;
  const _Float16* Kb = kf + (size_t)b * S_ * HD_;
  const _Float16* Vt = vt + (size_t)b * HD_ * S_;

  v16h qfr[8];
#pragma unroll
  for (int c = 0; c < 8; ++c) qfr[c] = load_a_frag(Q + c * 32, HD_, lane);

  v8f o[16];
#pragma unroll
  for (int t = 0; t < 16; ++t) { v8f z = {0,0,0,0,0,0,0,0}; o[t] = z; }
  float rmax[8], rsum[8];
#pragma unroll
  for (int r = 0; r < 8; ++r) { rmax[r] = -1e30f; rsum[r] = 0.f; }

  int jstart = qb0 - (WINDOW_ - 1);
  if (jstart < 0) jstart = 0;
  jstart &= ~31;
  const int jend = qb0 + 63;

  for (int kk0 = jstart; kk0 <= jend; kk0 += 32) {
    // ---- stage K chunk (32 keys x 256) and V chunk (256 d x 32 keys) ----
#if HAVE_TDM
    if (wave == 0) {
      tdm_load_2d((unsigned)(size_t)(void*)Klds, Kb + (size_t)kk0 * HD_,
                  (unsigned)HD_, 32u, (unsigned)HD_);
      tdm_load_2d((unsigned)(size_t)(void*)Vlds, Vt + kk0,
                  32u, (unsigned)HD_, (unsigned)S_);
      tdm_wait();
    }
#else
    for (int i = threadIdx.x * 8; i < 8192; i += 128 * 8)          // K contig
      *(v8h*)(Klds + i) = *(const v8h*)(Kb + (size_t)kk0 * HD_ + i);
    for (int d = threadIdx.x; d < 256; d += 128) {                 // V rows
      *(v8h*)(Vlds + d * 32)      = *(const v8h*)(Vt + (size_t)d * S_ + kk0);
      *(v8h*)(Vlds + d * 32 + 8)  = *(const v8h*)(Vt + (size_t)d * S_ + kk0 + 8);
      *(v8h*)(Vlds + d * 32 + 16) = *(const v8h*)(Vt + (size_t)d * S_ + kk0 + 16);
      *(v8h*)(Vlds + d * 32 + 24) = *(const v8h*)(Vt + (size_t)d * S_ + kk0 + 24);
    }
#endif
    __syncthreads();

    // ---- scores: two 16x16 C tiles, K = 256 in 8 chunks (from LDS) ----
    v8f s0 = {0,0,0,0,0,0,0,0};
    v8f s1 = {0,0,0,0,0,0,0,0};
#pragma unroll
    for (int c = 0; c < 8; ++c) {
      v16h b0 = load_b_frag(Klds + c * 32, HD_, lane);
      v16h b1 = load_b_frag(Klds + 16 * HD_ + c * 32, HD_, lane);
      s0 = wmma_f16(qfr[c], b0, s0);
      s1 = wmma_f16(qfr[c], b1, s1);
    }

    // ---- softcap + window mask + online softmax (masked lanes hard-zero) ----
    float sv0[8], sv1[8], pm[8];
    bool ok0[8], ok1[8];
#pragma unroll
    for (int r = 0; r < 8; ++r) {
      const int i  = qr0 + r + hi * 8;
      const int j0 = kk0 + ln;
      const int j1 = kk0 + 16 + ln;
      ok0[r] = (j0 <= i) && (j0 > i - WINDOW_);
      ok1[r] = (j1 <= i) && (j1 > i - WINDOW_);
      float x0 = SOFTCAP_ * tanhf(s0[r] * (SCALING_ / SOFTCAP_));
      float x1 = SOFTCAP_ * tanhf(s1[r] * (SCALING_ / SOFTCAP_));
      sv0[r] = ok0[r] ? x0 : -1e30f;
      sv1[r] = ok1[r] ? x1 : -1e30f;
      pm[r] = fmaxf(sv0[r], sv1[r]);
    }
#pragma unroll
    for (int xm = 1; xm < 16; xm <<= 1)
#pragma unroll
      for (int r = 0; r < 8; ++r) pm[r] = fmaxf(pm[r], __shfl_xor(pm[r], xm, 32));
    float alpha[8], psum[8];
#pragma unroll
    for (int r = 0; r < 8; ++r) {
      float mnew = fmaxf(rmax[r], pm[r]);
      alpha[r] = __expf(rmax[r] - mnew);
      rmax[r] = mnew;
      float p0 = ok0[r] ? __expf(sv0[r] - mnew) : 0.f;
      float p1 = ok1[r] ? __expf(sv1[r] - mnew) : 0.f;
      sv0[r] = p0; sv1[r] = p1;
      psum[r] = p0 + p1;
    }
#pragma unroll
    for (int xm = 1; xm < 16; xm <<= 1)
#pragma unroll
      for (int r = 0; r < 8; ++r) psum[r] += __shfl_xor(psum[r], xm, 32);
#pragma unroll
    for (int r = 0; r < 8; ++r) rsum[r] = rsum[r] * alpha[r] + psum[r];
#pragma unroll
    for (int t = 0; t < 16; ++t)
#pragma unroll
      for (int r = 0; r < 8; ++r) o[t][r] *= alpha[r];

    // ---- P: C-layout -> A-layout via per-wave LDS tile ----
#pragma unroll
    for (int r = 0; r < 8; ++r) {
      P[(r + hi * 8) * 32 + ln]      = (_Float16)sv0[r];
      P[(r + hi * 8) * 32 + 16 + ln] = (_Float16)sv1[r];
    }
    asm volatile("s_wait_dscnt 0" ::: "memory");
    v16h pf = load_a_frag(P, 32, lane);

    // ---- O += P @ V  (16 N-tiles over head dim, V from LDS) ----
#pragma unroll
    for (int t = 0; t < 16; ++t) {
      v16h bv = load_b_frag(Vlds + (size_t)(t * 16) * 32, 32, lane);
      o[t] = wmma_f16(pf, bv, o[t]);
    }
    __syncthreads();   // all waves done with K/V chunk before next TDM
  }

  float inv[8];
#pragma unroll
  for (int r = 0; r < 8; ++r) inv[r] = 1.0f / rsum[r];
#pragma unroll
  for (int t = 0; t < 16; ++t)
#pragma unroll
    for (int r = 0; r < 8; ++r) {
      const size_t m = (size_t)(qr0 + r + hi * 8);
      ao[((size_t)b * S_ + m) * NQD_ + h * HD_ + t * 16 + ln] =
          (_Float16)(o[t][r] * inv[r]);
    }
}

// ---------------------------------------------------------------------------
extern "C" void kernel_launch(void* const* d_in, const int* in_sizes, int n_in,
                              void* d_out, int out_size, void* d_ws, size_t ws_size,
                              hipStream_t stream) {
  const float* hidden = (const float*)d_in[0];
  const float* cosb   = (const float*)d_in[1];
  const float* sinb   = (const float*)d_in[2];
  // d_in[3] = attention_mask (recomputed analytically on device)
  const float* Wq = (const float*)d_in[4];
  const float* Wk = (const float*)d_in[5];
  const float* Wv = (const float*)d_in[6];
  const float* Wo = (const float*)d_in[7];
  const float* qw = (const float*)d_in[8];
  const float* kw = (const float*)d_in[9];

  char* ws = (char*)d_ws;
  size_t off = 0;
  auto take = [&](size_t bytes) -> char* {
    char* p = ws + off;
    off = (off + bytes + 255) & ~(size_t)255;
    return p;
  };
  _Float16* hf    = (_Float16*)take((size_t)B_ * S_ * HID_ * 2);
  _Float16* wqkvt = (_Float16*)take((size_t)QKVN_ * HID_ * 2);
  _Float16* wot   = (_Float16*)take((size_t)HID_ * NQD_ * 2);
  float*    qkv   = (float*)   take((size_t)B_ * S_ * QKVN_ * 4);
  _Float16* qf    = (_Float16*)take((size_t)B_ * NH_ * S_ * HD_ * 2);
  _Float16* kfp   = (_Float16*)take((size_t)B_ * S_ * HD_ * 2);
  _Float16* vtp   = (_Float16*)take((size_t)B_ * HD_ * S_ * 2);
  _Float16* ao    = (_Float16*)take((size_t)B_ * S_ * NQD_ * 2);

  cvt_f32_to_f16<<<dim3(2048), dim3(256), 0, stream>>>(hidden, hf, B_ * S_ * HID_);
  pack_wqkv<<<dim3(960), dim3(256), 0, stream>>>(Wq, Wk, Wv, wqkvt);
  pack_wo<<<dim3(640), dim3(256), 0, stream>>>(Wo, wot);

  // qkv = hidden @ [Wq|Wk|Wv] : M=8192, N=1536, K=640
  gemm16_kernel<<<dim3((B_ * S_) / 16, QKVN_ / 128), dim3(128), 0, stream>>>(
      hf, wqkvt, HID_, qkv, QKVN_);

  qkv_post_kernel<<<dim3(B_ * S_), dim3(192), 0, stream>>>(
      qkv, cosb, sinb, qw, kw, qf, kfp, vtp);

  attn_kernel<<<dim3(B_ * NH_ * (S_ / 64)), dim3(128), 0, stream>>>(
      qf, kfp, vtp, ao);

  // out = attn @ Wo : M=8192, N=640, K=1024
  gemm16_kernel<<<dim3((B_ * S_) / 16, HID_ / 128), dim3(128), 0, stream>>>(
      ao, wot, NQD_, (float*)d_out, HID_);
}
